// Linear_86139864088812
// MI455X (gfx1250) — 8-bit matrix operand order repaired and checked by reading, compiled, not run on hardware
//
#include <hip/hip_runtime.h>
#include <hip/hip_bf16.h>

#define M_ 4096
#define N_ 16384
#define K_ 7168
#define KB (K_ / 128)

#define BSTRIDE 144
#define BUFSZ (128 * BSTRIDE)

typedef __attribute__((ext_vector_type(16))) int   v16i;
typedef __attribute__((ext_vector_type(8)))  float v8f;

static __device__ __forceinline__ unsigned int f32_to_e4m3(float f) {
    unsigned int u   = __float_as_uint(f);
    unsigned int sgn = (u >> 24) & 0x80u;
    float ax = fabsf(f);
    unsigned int au = u & 0x7fffffffu;
    unsigned int r;
    if (!(ax < 448.0f)) {
        r = 0x7eu;
    } else if (ax < 0.015625f) {
        int k = (int)rintf(ax * 512.0f);
        r = (unsigned int)k;
    } else {
        unsigned int lsb = (au >> 20) & 1u;
        unsigned int v   = au + 0x0007FFFFu + lsb;
        int exp          = (int)(v >> 23) - 127;
        unsigned int man = (v >> 20) & 7u;
        if (exp > 8 || (exp == 8 && man > 6u)) r = 0x7eu;
        else r = (unsigned int)(((exp + 7) << 3) | man);
    }
    return sgn | r;
}

static __device__ __forceinline__ unsigned int pack4_e4m3(float a, float b,
                                                          float c, float d) {
    return  f32_to_e4m3(a)        |
           (f32_to_e4m3(b) <<  8) |
           (f32_to_e4m3(c) << 16) |
           (f32_to_e4m3(d) << 24);
}

static __device__ __forceinline__ void wait_async0() {
#if __has_builtin(__builtin_amdgcn_s_wait_asynccnt)
    __builtin_amdgcn_s_wait_asynccnt(0);
#else
    asm volatile("s_wait_asynccnt 0x0" ::: "memory");
#endif
}

__global__ __launch_bounds__(256) void quant_x_kernel(
    const float* __restrict__ x, unsigned char* __restrict__ xq,
    float* __restrict__ xs_t) {
    const int lane = threadIdx.x & 31;
    const int blk  = blockIdx.x * (blockDim.x >> 5) + (threadIdx.x >> 5);
    if (blk >= M_ * KB) return;
    const int m  = blk / KB;
    const int kb = blk % KB;

    const float4 v = *(const float4*)(x + (size_t)m * K_ + kb * 128 + lane * 4);
    float amax = fmaxf(fmaxf(fabsf(v.x), fabsf(v.y)),
                       fmaxf(fabsf(v.z), fabsf(v.w)));
#pragma unroll
    for (int off = 16; off >= 1; off >>= 1)
        amax = fmaxf(amax, __shfl_xor(amax, off, 32));

    const float s   = amax * (1.0f / 448.0f);
    const float inv = (s > 0.0f) ? (1.0f / s) : 0.0f;

    const unsigned int packed =
        pack4_e4m3(v.x * inv, v.y * inv, v.z * inv, v.w * inv);
    *(unsigned int*)(xq + (size_t)m * K_ + kb * 128 + lane * 4) = packed;
    if (lane == 0) xs_t[(size_t)kb * M_ + m] = s;
}

__global__ __launch_bounds__(256) void quant_w_kernel(
    const float* __restrict__ w, unsigned char* __restrict__ wq,
    long long total4) {
    long long i = (long long)blockIdx.x * blockDim.x + threadIdx.x;
    if (i >= total4) return;
    const float4 v = ((const float4*)w)[i];
    ((unsigned int*)wq)[i] = pack4_e4m3(v.x, v.y, v.z, v.w);
}

__global__ __launch_bounds__(256) void fp8_gemm_kernel(
    const unsigned char* __restrict__ xq, const float* __restrict__ xs_t,
    const unsigned char* __restrict__ wq, const float* __restrict__ wscale,
    float* __restrict__ out) {
    __shared__ unsigned char smem[2 * BUFSZ];

    const int lane    = threadIdx.x & 31;
    const int wave    = threadIdx.x >> 5;
    const int tileN   = blockIdx.x * 128;
    const int tileM   = blockIdx.y * 128;
    const int mBase   = tileM + wave * 16;
    const int halfSel = lane >> 4;
    const int lmod    = lane & 15;

    const unsigned char* aRow = xq + (size_t)(mBase + lmod) * K_ + halfSel * 16;
    const float* wsRow = wscale + (size_t)(tileN >> 7) * KB;

    const unsigned ldsBase = (unsigned)(unsigned long long)&smem[0];

    auto stage = [&](int kb, int buf) {
#pragma unroll
        for (int i = 0; i < 4; ++i) {
            const int c   = (int)threadIdx.x + i * 256;
            const int row = c >> 3;
            const int col = c & 7;
            const unsigned long long g = (unsigned long long)(
                wq + (size_t)(tileN + row) * K_ + (size_t)kb * 128 + col * 16);
            const unsigned l = ldsBase + buf * BUFSZ + row * BSTRIDE + col * 16;
            asm volatile("global_load_async_to_lds_b128 %0, %1, off"
                         :: "v"(l), "v"(g) : "memory");
        }
    };

    v8f acc[8];
#pragma unroll
    for (int i = 0; i < 8; ++i) acc[i] = (v8f){0, 0, 0, 0, 0, 0, 0, 0};

    stage(0, 0);

#pragma unroll 1
    for (int kb = 0; kb < KB; ++kb) {
        const int buf = kb & 1;
        wait_async0();
        __syncthreads();

        if (kb + 1 < KB) stage(kb + 1, buf ^ 1);

        v16i a;
#pragma unroll
        for (int p = 0; p < 4; ++p) {
            uint4 d = *(const uint4*)(aRow + (size_t)kb * 128 + p * 32);
            a[4 * p]     = (int)d.x;
            a[4 * p + 1] = (int)d.y;
            a[4 * p + 2] = (int)d.z;
            a[4 * p + 3] = (int)d.w;
        }

        const float  ws    = wsRow[kb];
        const float* xsRow = xs_t + (size_t)kb * M_ + mBase + halfSel * 8;
        const float4 s0    = *(const float4*)xsRow;
        const float4 s1    = *(const float4*)(xsRow + 4);
        float sc[8] = {s0.x * ws, s0.y * ws, s0.z * ws, s0.w * ws,
                       s1.x * ws, s1.y * ws, s1.z * ws, s1.w * ws};

        const unsigned char* bTile = smem + buf * BUFSZ;
#pragma unroll
        for (int ns = 0; ns < 8; ++ns) {
            const unsigned char* bRow =
                bTile + (ns * 16 + lmod) * BSTRIDE + halfSel * 16;
            v16i b;
#pragma unroll
            for (int c = 0; c < 4; ++c) {
                int4 d = *(const int4*)(bRow + c * 32);
                b[4 * c + 0] = d.x;
                b[4 * c + 1] = d.y;
                b[4 * c + 2] = d.z;
                b[4 * c + 3] = d.w;
            }
            v8f cz = (v8f){0, 0, 0, 0, 0, 0, 0, 0};
            v8f p8 = __builtin_amdgcn_wmma_f32_16x16x128_fp8_fp8(
                a, b, (short)0, cz, false, false);
#pragma unroll
            for (int i = 0; i < 8; ++i)
                acc[ns][i] = fmaf(p8[i], sc[i], acc[ns][i]);
        }
    }

#pragma unroll
    for (int ns = 0; ns < 8; ++ns) {
#pragma unroll
        for (int i = 0; i < 8; ++i) {
            const size_t m = (size_t)(mBase + halfSel * 8 + i);
            out[m * N_ + tileN + ns * 16 + lmod] = acc[ns][i];
        }
    }
}

extern "C" void kernel_launch(void* const* d_in, const int* in_sizes, int n_in,
                              void* d_out, int out_size, void* d_ws,
                              size_t ws_size, hipStream_t stream) {
    const float* x      = (const float*)d_in[0];
    const float* w      = (const float*)d_in[1];
    const float* wscale = (const float*)d_in[2];
    (void)in_sizes; (void)n_in; (void)out_size; (void)ws_size;

    unsigned char* base = (unsigned char*)d_ws;
    unsigned char* xq   = base;
    size_t off = ((size_t)M_ * K_ + 255) & ~(size_t)255;
    float* xs_t = (float*)(base + off);
    off += (((size_t)KB * M_ * sizeof(float)) + 255) & ~(size_t)255;
    unsigned char* wq = base + off;

    const int nblk = M_ * KB;
    quant_x_kernel<<<(nblk + 7) / 8, 256, 0, stream>>>(x, xq, xs_t);

    const long long total4 = (long long)N_ * K_ / 4;
    quant_w_kernel<<<(unsigned)((total4 + 255) / 256), 256, 0, stream>>>(
        w, wq, total4);

    dim3 grid(N_ / 128, M_ / 128);
    fp8_gemm_kernel<<<grid, 256, 0, stream>>>(xq, xs_t, wq, wscale,
                                              (float*)d_out);
}
